// _AWQ4Linear_64965675319552
// MI455X (gfx1250) — matrix operand order repaired and checked by reading, not compiled, not run on hardware
//
#include <hip/hip_runtime.h>

// ---- CDNA5 (gfx1250) wave32 WMMA types ----
typedef __attribute__((ext_vector_type(16))) __bf16 v16bf;
typedef __attribute__((ext_vector_type(8)))  __bf16 v8bf;
typedef __attribute__((ext_vector_type(8)))  float  v8f;
typedef __attribute__((ext_vector_type(4)))  float  v4f;
typedef __attribute__((ext_vector_type(4)))  int    v4i;

#define BM 128           // M tile per workgroup
#define BN 128           // N tile per workgroup
#define KT 32            // K step (== WMMA K for bf16)
#define LDSK 40          // padded LDS row length in bf16 elems (80 B, bank-conflict-free, 16B aligned)

__global__ __launch_bounds__(256)
void awq4_wmma_bf16(const float* __restrict__ x,        // [M, K] fp32
                    const int*   __restrict__ wq,       // [N, K/2] int32, one packed byte each
                    const float* __restrict__ scales,   // [N, n_g]
                    const float* __restrict__ zeros,    // [N, n_g]
                    const float* __restrict__ col_scale,// [K]
                    const float* __restrict__ bias,     // [N]
                    float* __restrict__ out,            // [M, N]
                    int M, int K, int N, int n_g, int gs)
{
  __shared__ __align__(16) __bf16 sA[2][BM * LDSK];   // x tile, bf16, [m][k]
  __shared__ __align__(16) __bf16 sB[2][BN * LDSK];   // dequantized w tile, bf16, [n][k]

  const int tid  = threadIdx.x;
  const int lane = tid & 31;
  const int wave = tid >> 5;       // 8 waves
  const int wm   = wave & 1;       // which 64-row M half
  const int wn   = wave >> 1;      // which 32-col N quarter
  const int l16  = lane & 15;
  const int kh   = lane >> 4;      // lane half (K-half selector in fragments)

  const int mBase = blockIdx.y * BM;
  const int nBase = blockIdx.x * BN;

  // staging assignment: 2 threads per tile row, each covers a 16-wide K half
  const int srow  = tid >> 1;      // 0..127
  const int shalf = tid & 1;       // 0..1

  v8f acc[4][2];
  #pragma unroll
  for (int i = 0; i < 4; ++i)
    #pragma unroll
    for (int j = 0; j < 2; ++j)
      #pragma unroll
      for (int r = 0; r < 8; ++r) acc[i][j][r] = 0.0f;

  const long xrow  = (long)(mBase + srow) * K;
  const long qrow  = (long)(nBase + srow) * (K >> 1);
  const long scrow = (long)(nBase + srow) * n_g;

  // ---- wave-level WMMA on one staged K-tile ----
  auto compute = [&](int buf) {
    // B fragments (32x16 bf16): col n = l16, per-lane K = kh*16 + e  (e = 0..15)
    v16bf bf[2];
    #pragma unroll
    for (int nt = 0; nt < 2; ++nt) {
      const __bf16* p = &sB[buf][(wn * 32 + nt * 16 + l16) * LDSK + kh * 16];
      v8bf b0 = *(const v8bf*)(p);
      v8bf b1 = *(const v8bf*)(p + 8);
      bf[nt] = __builtin_shufflevector(b0, b1, 0,1,2,3,4,5,6,7,8,9,10,11,12,13,14,15);
    }
    // A fragments (16x32 bf16): row m = l16, per-lane K = kh*16 + e  (e = 0..15)
    #pragma unroll
    for (int mt = 0; mt < 4; ++mt) {
      const __bf16* p = &sA[buf][(wm * 64 + mt * 16 + l16) * LDSK];
      v8bf a0 = *(const v8bf*)(p + kh * 16);
      v8bf a1 = *(const v8bf*)(p + kh * 16 + 8);
      v16bf af = __builtin_shufflevector(a0, a1, 0,1,2,3,4,5,6,7,8,9,10,11,12,13,14,15);
      #pragma unroll
      for (int nt = 0; nt < 2; ++nt) {
        // (neg_a, A, neg_b, B, c_mod, C, reuse_a, reuse_b)
        acc[mt][nt] = __builtin_amdgcn_wmma_f32_16x16x32_bf16(
            false, af, false, bf[nt], (short)0, acc[mt][nt], false, false);
      }
    }
  };

  const int nT = K / KT;
  for (int kt = 0; kt < nT; ++kt) {
    const int k0 = kt * KT;

    // ---- issue global loads for tile kt (kept in flight across compute) ----
    const float* xp = x + xrow + k0 + shalf * 16;
    v4f xv0 = *(const v4f*)(xp + 0);
    v4f xv1 = *(const v4f*)(xp + 4);
    v4f xv2 = *(const v4f*)(xp + 8);
    v4f xv3 = *(const v4f*)(xp + 12);

    const int* qp = wq + qrow + (k0 >> 1) + shalf * 8;
    v4i q0 = *(const v4i*)(qp + 0);
    v4i q1 = *(const v4i*)(qp + 4);

    const int g   = k0 / gs;                 // KT=32 lies inside one quant group
    const float sc = scales[scrow + g];
    const float zp = zeros [scrow + g];

    const float* cs = col_scale + k0 + shalf * 16;
    v4f c0 = *(const v4f*)(cs + 0);
    v4f c1 = *(const v4f*)(cs + 4);
    v4f c2 = *(const v4f*)(cs + 8);
    v4f c3 = *(const v4f*)(cs + 12);

    if (kt + 4 < nT) {                       // L2 prefetch a few K-tiles ahead
      __builtin_prefetch(xp + 4 * KT, 0, 1);
      __builtin_prefetch(qp + 4 * (KT >> 1), 0, 1);
    }

    // ---- overlap: WMMA on previous tile while the loads are outstanding ----
    if (kt) compute((kt - 1) & 1);

    __syncthreads();

    const int buf = kt & 1;
    // stage x -> bf16 LDS
    {
      v8bf a0, a1;
      #pragma unroll
      for (int j = 0; j < 4; ++j) { a0[j] = (__bf16)xv0[j]; a0[4 + j] = (__bf16)xv1[j]; }
      #pragma unroll
      for (int j = 0; j < 4; ++j) { a1[j] = (__bf16)xv2[j]; a1[4 + j] = (__bf16)xv3[j]; }
      __bf16* dst = &sA[buf][srow * LDSK + shalf * 16];
      *(v8bf*)(dst + 0) = a0;
      *(v8bf*)(dst + 8) = a1;
    }
    // dequantize weights -> bf16 LDS   (even col = lo nibble, odd col = hi nibble)
    {
      float cf[16];
      *(v4f*)(cf + 0)  = c0; *(v4f*)(cf + 4)  = c1;
      *(v4f*)(cf + 8)  = c2; *(v4f*)(cf + 12) = c3;
      v8bf w0, w1;
      #pragma unroll
      for (int j = 0; j < 4; ++j) {
        int v = q0[j];
        w0[2 * j]     = (__bf16)(fmaf(sc, (float)( v        & 15), zp) * cf[2 * j]);
        w0[2 * j + 1] = (__bf16)(fmaf(sc, (float)((v >> 4)  & 15), zp) * cf[2 * j + 1]);
      }
      #pragma unroll
      for (int j = 0; j < 4; ++j) {
        int v = q1[j];
        w1[2 * j]     = (__bf16)(fmaf(sc, (float)( v        & 15), zp) * cf[8 + 2 * j]);
        w1[2 * j + 1] = (__bf16)(fmaf(sc, (float)((v >> 4)  & 15), zp) * cf[8 + 2 * j + 1]);
      }
      __bf16* dst = &sB[buf][srow * LDSK + shalf * 16];
      *(v8bf*)(dst + 0) = w0;
      *(v8bf*)(dst + 8) = w1;
    }
    __syncthreads();
  }
  compute((nT - 1) & 1);

  // ---- epilogue: bias + store (C layout: VGPR r -> row kh*8 + r, col = l16) ----
  #pragma unroll
  for (int nt = 0; nt < 2; ++nt) {
    const int col = nBase + wn * 32 + nt * 16 + l16;
    const float bv = bias[col];
    #pragma unroll
    for (int mt = 0; mt < 4; ++mt) {
      const int rowb = mBase + wm * 64 + mt * 16 + kh * 8;
      #pragma unroll
      for (int r = 0; r < 8; ++r) {
        out[(long)(rowb + r) * N + col] = acc[mt][nt][r] + bv;
      }
    }
  }
}

extern "C" void kernel_launch(void* const* d_in, const int* in_sizes, int n_in,
                              void* d_out, int out_size, void* d_ws, size_t ws_size,
                              hipStream_t stream) {
  const float* x         = (const float*)d_in[0];
  const int*   wq        = (const int*)  d_in[1];
  const float* scales    = (const float*)d_in[2];
  const float* zeros     = (const float*)d_in[3];
  const float* col_scale = (const float*)d_in[4];
  const float* bias      = (const float*)d_in[5];
  float*       out       = (float*)d_out;

  const int I   = in_sizes[4];          // 4096
  const int O   = in_sizes[5];          // 11008
  const int M   = in_sizes[0] / I;      // 8192
  const int n_g = in_sizes[2] / O;      // 64
  const int gs  = I / n_g;              // 64

  dim3 grid(O / BN, M / BM);            // 86 x 64
  awq4_wmma_bf16<<<grid, 256, 0, stream>>>(x, wq, scales, zeros, col_scale, bias, out,
                                           M, I, O, n_g, gs);
}
